// RNN_Cell_62835371540839
// MI455X (gfx1250) — matrix operand order repaired and checked by reading, not compiled, not run on hardware
//
#include <hip/hip_runtime.h>
#include <hip/hip_bf16.h>
#include <math.h>

// Problem constants (from reference)
#define NCH   4
#define NHID  2048
#define NTOT  (NCH * NHID)   // 8192 output columns
#define KDIM  2048           // reduction length

// Tiling
#define BM 64
#define BN 256
#define BK 32
#define NK (KDIM / BK)       // 64 K-steps
#define LDT (BK + 8)         // bf16 LDS stride (40 elems = 80B) -> bank spread

typedef __attribute__((ext_vector_type(16))) __bf16 v16bf;
typedef __attribute__((ext_vector_type(8)))  __bf16 v8bf;
typedef __attribute__((ext_vector_type(8)))  float  v8f;

__device__ __forceinline__ float4 ld4(const float* p) { return *(const float4*)p; }

__device__ __forceinline__ v8bf cvt8(float4 a, float4 b) {
  v8bf o;
  o[0] = (__bf16)a.x; o[1] = (__bf16)a.y; o[2] = (__bf16)a.z; o[3] = (__bf16)a.w;
  o[4] = (__bf16)b.x; o[5] = (__bf16)b.y; o[6] = (__bf16)b.z; o[7] = (__bf16)b.w;
  return o;
}

__device__ __forceinline__ v16bf combine(v8bf lo, v8bf hi) {
  return __builtin_shufflevector(lo, hi, 0, 1, 2, 3, 4, 5, 6, 7,
                                         8, 9, 10, 11, 12, 13, 14, 15);
}

__global__ __launch_bounds__(256)
void rnn_cell_wmma_kernel(const float* __restrict__ x,
                          const float* __restrict__ hy,
                          const float* __restrict__ Wxw,
                          const float* __restrict__ Wxb,
                          const float* __restrict__ Wyw,
                          const float* __restrict__ Wyb,
                          float* __restrict__ out) {
  // Double-buffered bf16 tiles
  __shared__ __bf16 sA[2][BM * LDT];   // A: 64 rows  x 32 K
  __shared__ __bf16 sB[2][BN * LDT];   // B: 256 rows x 32 K

  const int tid  = threadIdx.x;
  const int wave = tid >> 5;
  const int lane = tid & 31;
  const int half = lane >> 4;
  const int lcol = lane & 15;
  const int wm   = wave >> 2;  // 0..1 -> 32-row slab
  const int wn   = wave & 3;   // 0..3 -> 64-col slab

  const int bm0 = blockIdx.y * BM;
  const int bn0 = blockIdx.x * BN;

  // Staging assignment: each thread owns one 8-f32 chunk of A and four of B.
  // chunk c: row = c>>2, kc = (c&3)*8 ; B chunk i lives at row arow + 64*i.
  const int arow = tid >> 2;
  const int akc  = (tid & 3) << 3;
  const float* gA = x   + (size_t)(bm0 + arow) * KDIM + akc;
  const float* gB = Wxw + (size_t)(bn0 + arow) * KDIM + akc;

  v8f acc[2][4];
#pragma unroll
  for (int i = 0; i < 2; ++i)
#pragma unroll
    for (int j = 0; j < 4; ++j)
#pragma unroll
      for (int r = 0; r < 8; ++r) acc[i][j][r] = 0.0f;

  float4 ra0, ra1, rb0[4], rb1[4];

  // ---- prologue: load + convert + store tile 0
  ra0 = ld4(gA);
  ra1 = ld4(gA + 4);
#pragma unroll
  for (int i = 0; i < 4; ++i) {
    rb0[i] = ld4(gB + (size_t)(i * 64) * KDIM);
    rb1[i] = ld4(gB + (size_t)(i * 64) * KDIM + 4);
  }
  *(v8bf*)&sA[0][arow * LDT + akc] = cvt8(ra0, ra1);
#pragma unroll
  for (int i = 0; i < 4; ++i)
    *(v8bf*)&sB[0][(arow + i * 64) * LDT + akc] = cvt8(rb0[i], rb1[i]);
  __syncthreads();

  for (int ks = 0; ks < NK; ++ks) {
    const int cur = ks & 1;

    // ---- issue global loads for tile ks+1 (latency hidden by compute)
    if (ks + 1 < NK) {
      const int k1 = (ks + 1) * BK;
      ra0 = ld4(gA + k1);
      ra1 = ld4(gA + k1 + 4);
#pragma unroll
      for (int i = 0; i < 4; ++i) {
        rb0[i] = ld4(gB + (size_t)(i * 64) * KDIM + k1);
        rb1[i] = ld4(gB + (size_t)(i * 64) * KDIM + k1 + 4);
      }
    }

    // ---- compute from buffer `cur`: 12 ds_load_b128 + 8 WMMA
    const __bf16* A = sA[cur];
    const __bf16* Bt = sB[cur];

    v16bf afrag[2];
#pragma unroll
    for (int mf = 0; mf < 2; ++mf) {
      const __bf16* p = A + (wm * 32 + mf * 16 + lcol) * LDT;
      v8bf c0 = *(const v8bf*)(p + half * 16);
      v8bf c1 = *(const v8bf*)(p + half * 16 + 8);
      afrag[mf] = combine(c0, c1);
    }
    v16bf bfrag[4];
#pragma unroll
    for (int nf = 0; nf < 4; ++nf) {
      const __bf16* p = Bt + (wn * 64 + nf * 16 + lcol) * LDT + half * 16;
      v8bf c0 = *(const v8bf*)(p);
      v8bf c1 = *(const v8bf*)(p + 8);
      bfrag[nf] = combine(c0, c1);
    }

#pragma unroll
    for (int mf = 0; mf < 2; ++mf)
#pragma unroll
      for (int nf = 0; nf < 4; ++nf)
        acc[mf][nf] = __builtin_amdgcn_wmma_f32_16x16x32_bf16(
            false, afrag[mf], false, bfrag[nf], (short)0, acc[mf][nf],
            false, false);

    // ---- convert + store tile ks+1 into the other buffer
    if (ks + 1 < NK) {
      const int nxt = cur ^ 1;
      *(v8bf*)&sA[nxt][arow * LDT + akc] = cvt8(ra0, ra1);
#pragma unroll
      for (int i = 0; i < 4; ++i)
        *(v8bf*)&sB[nxt][(arow + i * 64) * LDT + akc] = cvt8(rb0[i], rb1[i]);
    }
    __syncthreads();
  }

  // ---- epilogue: + Wx_b + locally-connected(h) + Wy_b, tanh, store
  // D layout: VGPR r -> M = r + 8*half ; N = lane&15
#pragma unroll
  for (int mf = 0; mf < 2; ++mf) {
#pragma unroll
    for (int nf = 0; nf < 4; ++nf) {
#pragma unroll
      for (int r = 0; r < 8; ++r) {
        const int m = bm0 + wm * 32 + mf * 16 + 8 * half + r;
        const int n = bn0 + wn * 64 + nf * 16 + lcol;
        float v = acc[mf][nf][r] + Wxb[n];

        const int oc = n >> 11;       // n / NHID
        const int j  = n & (NHID - 1);
        const int jm = (j == 0) ? (NHID - 1) : j - 1;
        const int jp = (j == NHID - 1) ? 0 : j + 1;

        float ly = Wyb[(oc << 11) + j];
#pragma unroll
        for (int ic = 0; ic < NCH; ++ic) {
          const float* hrow = hy + (size_t)m * NTOT + (ic << 11);
          const float* w    = Wyw + ((size_t)((oc * NCH + ic) << 11) + j) * 3;
          ly = fmaf(hrow[jm], w[0], ly);
          ly = fmaf(hrow[j],  w[1], ly);
          ly = fmaf(hrow[jp], w[2], ly);
        }
        out[(size_t)m * NTOT + n] = tanhf(v + ly);
      }
    }
  }
}

extern "C" void kernel_launch(void* const* d_in, const int* in_sizes, int n_in,
                              void* d_out, int out_size, void* d_ws, size_t ws_size,
                              hipStream_t stream) {
  const float* x   = (const float*)d_in[0];
  const float* hy  = (const float*)d_in[1];
  const float* Wxw = (const float*)d_in[2];
  const float* Wxb = (const float*)d_in[3];
  const float* Wyw = (const float*)d_in[4];
  const float* Wyb = (const float*)d_in[5];
  float* out = (float*)d_out;

  const int B = in_sizes[0] / KDIM;  // 1024
  dim3 grid(NTOT / BN, B / BM);      // (32, 16)
  dim3 block(256);
  rnn_cell_wmma_kernel<<<grid, block, 0, stream>>>(x, hy, Wxw, Wxb, Wyw, Wyb, out);
}
